// MACE_76441827934638
// MI455X (gfx1250) — matrix operand order repaired and checked by reading, not compiled, not run on hardware
//
#include <hip/hip_runtime.h>
#include <hip/hip_bf16.h>

#define DEVINL __device__ __forceinline__

typedef __attribute__((ext_vector_type(16))) _Float16 v16h;
typedef __attribute__((ext_vector_type(8)))  _Float16 v8h;
typedef __attribute__((ext_vector_type(8)))  float    v8f;
typedef __attribute__((ext_vector_type(4)))  float    v4f;

constexpr int Nn = 8192;    // nodes
constexpr int Ee = 131072;  // edges
constexpr int Cc = 128;     // channels

// ---------------------------------------------------------------------------
// WMMA helpers (gfx1250 wave32, 16x16x32 f16 -> f32)
//
// A fragment (16x32, row-major source, K contiguous): lane L (h=L>>4, m=L&15)
//   halves 0..7  = A[m][kk + 8h .. kk + 8h + 7]        (one b128 load)
//   halves 8..15 = A[m][kk + 16 + 8h .. kk + 16+8h+7]  (one b128 load)
// B fragment (32x16, row-major [K][N] source): lane L (h=L>>4, n=L&15) holds column n0 + n,
//   halves 0..7 = B[kk+8h+i][n0+n], halves 8..15 = B[kk+16+8h+i][n0+n] (strided loads)
// C/D: lane L (h=L>>4, n=L&15), vgpr r -> D[r + 8h][n]
// ---------------------------------------------------------------------------

DEVINL v8f wmma32(v16h a, v16h b, v8f c) {
  return __builtin_amdgcn_wmma_f32_16x16x32_f16(false, a, false, b, (short)0, c,
                                                false, false);
}

DEVINL v16h frag_a_h(const _Float16* A, int lda, int kk, int lane) {
  const int hh = (lane >> 4) & 1, m = lane & 15;
  const _Float16* p = A + (size_t)m * lda + kk + hh * 8;
  v8h lo = *(const v8h*)(p);
  v8h hi = *(const v8h*)(p + 16);
  v16h a;
#pragma unroll
  for (int i = 0; i < 8; ++i) { a[i] = lo[i]; a[i + 8] = hi[i]; }
  return a;
}

// A fragment loaded from an f32 source, converted on the fly to f16.
DEVINL v16h frag_a_f(const float* A, int lda, int kk, int lane) {
  const int hh = (lane >> 4) & 1, m = lane & 15;
  const float* p = A + (size_t)m * lda + kk + hh * 8;
  v4f x0 = *(const v4f*)(p);
  v4f x1 = *(const v4f*)(p + 4);
  v4f y0 = *(const v4f*)(p + 16);
  v4f y1 = *(const v4f*)(p + 20);
  v16h a;
#pragma unroll
  for (int i = 0; i < 4; ++i) {
    a[i]      = (_Float16)x0[i];
    a[4 + i]  = (_Float16)x1[i];
    a[8 + i]  = (_Float16)y0[i];
    a[12 + i] = (_Float16)y1[i];
  }
  return a;
}

DEVINL v16h frag_b_h(const _Float16* B, int ldb, int n0, int kk, int lane) {
  const int hh = (lane >> 4) & 1, n = lane & 15;
  const _Float16* p = B + (size_t)(kk + hh * 8) * ldb + n0 + n;
  v16h b;
#pragma unroll
  for (int i = 0; i < 8; ++i) { b[i] = p[(size_t)i * ldb]; b[i + 8] = p[(size_t)(16 + i) * ldb]; }
  return b;
}

DEVINL float silu(float x) { return x / (1.0f + __expf(-x)); }

// CDNA5 async memory->LDS copy (bypasses VGPRs, tracked by ASYNCcnt).
// lds_addr is the byte offset within the workgroup's LDS allocation.
DEVINL void async_ld_b128(unsigned lds_addr, const void* gaddr) {
  asm volatile("global_load_async_to_lds_b128 %0, %1, off"
               :
               : "v"(lds_addr), "v"((unsigned long long)(uintptr_t)gaddr)
               : "memory");
}
DEVINL void wait_async0() {
  asm volatile("s_wait_asynccnt 0x0" ::: "memory");
}

// ---------------------------------------------------------------------------
// Utility kernels
// ---------------------------------------------------------------------------
__global__ void k_zero(float* p, size_t n) {
  size_t i = (size_t)blockIdx.x * blockDim.x + threadIdx.x;
  if (i < n) p[i] = 0.0f;
}

__global__ void k_cvt(const float* __restrict__ s, _Float16* __restrict__ d, int n) {
  int i = blockIdx.x * blockDim.x + threadIdx.x;
  if (i < n) d[i] = (_Float16)s[i];
}

// Wr1 [8][64] -> padded f16 [32][64] (rows 8..31 zero) so K=32 for layer 1.
__global__ void k_wr1pad(const float* __restrict__ s, _Float16* __restrict__ d) {
  int i = blockIdx.x * blockDim.x + threadIdx.x;
  if (i < 32 * 64) {
    int r = i >> 6, c = i & 63;
    d[i] = (_Float16)(r < 8 ? s[r * 64 + c] : 0.0f);
  }
}

// ---------------------------------------------------------------------------
// Node init: elem = argmax(one-hot), h0 = W_embed[elem] (f16 for GEMM),
// sc0[n,o] = sum_k h0[k] * W_skip[k, elem, o]   (per-element skip matrix)
// ---------------------------------------------------------------------------
__global__ __launch_bounds__(128) void k_node_init(
    const float* __restrict__ x, const float* __restrict__ Wemb,
    const float* __restrict__ Wskip, int* __restrict__ elem,
    _Float16* __restrict__ h0h, float* __restrict__ sc0) {
  __shared__ float sh[128];
  __shared__ int se;
  const int n = blockIdx.x, c = threadIdx.x;
  if (c == 0) {
    int e = 0;
    float best = x[(size_t)n * 10];
    for (int a = 1; a < 10; ++a) {
      float v = x[(size_t)n * 10 + a];
      if (v > best) { best = v; e = a; }
    }
    se = e;
    elem[n] = e;
  }
  __syncthreads();
  const int e = se;
  float h0 = Wemb[e * Cc + c];
  h0h[(size_t)n * Cc + c] = (_Float16)h0;
  sh[c] = h0;
  __syncthreads();
  float acc = 0.0f;
  for (int k = 0; k < Cc; ++k)
    acc += sh[k] * Wskip[((size_t)k * 10 + e) * Cc + c];
  sc0[(size_t)n * Cc + c] = acc;
}

// ---------------------------------------------------------------------------
// h = h0 @ W_up   (WMMA, f32 out since h feeds f32 message math)
// ---------------------------------------------------------------------------
__global__ __launch_bounds__(32) void k_gemm_h(
    const _Float16* __restrict__ h0h, const _Float16* __restrict__ Wuph,
    float* __restrict__ hfeat) {
  const int lane = threadIdx.x;
  const size_t m0 = (size_t)blockIdx.x * 16;
  const int n0 = blockIdx.y * 16;
  v8f acc = {};
#pragma unroll
  for (int kk = 0; kk < 128; kk += 32) {
    v16h a = frag_a_h(h0h + m0 * Cc, Cc, kk, lane);
    v16h b = frag_b_h(Wuph, Cc, n0, kk, lane);
    acc = wmma32(a, b, acc);
  }
  const int hh = lane >> 4, nn = lane & 15;
#pragma unroll
  for (int rr = 0; rr < 8; ++rr)
    hfeat[(m0 + rr + 8 * hh) * Cc + n0 + nn] = acc[rr];
}

// ---------------------------------------------------------------------------
// Fused edge kernel: one wave per 16 edges.
//   async DMA of h[sender] into LDS overlapped with inline SH + Bessel,
//   4-layer MLP (WMMA, LDS ping-pong),
//   msg = Rw[LMAP] * Y * h[sender] / 16 -> atomic scatter to agg[N,16,128].
// ---------------------------------------------------------------------------
struct EdgeSmem {
  _Float16 act0[4][16][64];
  _Float16 act1[4][16][64];
  float Yt[4][16][16];
  float hs[4][16][128];
  int rcv[4][16];
};

__global__ __launch_bounds__(128) void k_edge(
    const float* __restrict__ ev, const float* __restrict__ elen,
    const int* __restrict__ eidx, const float* __restrict__ hfeat,
    const _Float16* __restrict__ Wr1h, const _Float16* __restrict__ Wr2h,
    const _Float16* __restrict__ Wr3h, const _Float16* __restrict__ Wr4h,
    float* agg) {
  __shared__ EdgeSmem sm;

  const int w = threadIdx.x >> 5;
  const int lane = threadIdx.x & 31;
  const int hh = lane >> 4, nn = lane & 15;
  const size_t e0 = ((size_t)blockIdx.x * 4 + w) * 16;

  // Warm WGP$/L2 for the big layer-4 weight block (64 KB, streamed per wave).
  __builtin_prefetch(Wr4h + (size_t)threadIdx.x * 256, 0, 3);

  // Issue the sender-feature DMA first: 16 rows x 512 B, memory -> LDS with
  // no VGPR round-trip; ASYNCcnt tracks completion while we do VALU-heavy
  // spherical harmonics / Bessel math below.
#pragma unroll
  for (int t = 0; t < 16; ++t) {
    int snd = eidx[e0 + t];  // sender row (uniform -> scalarized)
    unsigned lds_addr =
        (unsigned)(__builtin_offsetof(EdgeSmem, hs) +
                   (((w * 16 + t) * 128 + lane * 4) * sizeof(float)));
    async_ld_b128(lds_addr, hfeat + (size_t)snd * Cc + lane * 4);
  }

  if (lane < 16) {
    const size_t e = e0 + lane;
    float vx = ev[e * 3 + 0], vy = ev[e * 3 + 1], vz = ev[e * 3 + 2];
    float r = elen[e];
    float rn = rsqrtf(vx * vx + vy * vy + vz * vz);
    float X = vx * rn, Yy = vy * rn, Z = vz * rn;
    const float s3 = 1.7320508075688772f, s5 = 2.2360679774997896f;
    const float s15 = 3.872983346207417f, s70_4 = 2.091650066335189f;
    const float s105 = 10.246950765959598f, s105_2 = 5.123475382979799f;
    const float s42_4 = 1.6201851746019651f, s7_2 = 1.3228756555322954f;
    float* Yw = &sm.Yt[w][lane][0];
    Yw[0] = 1.0f;
    Yw[1] = s3 * X;  Yw[2] = s3 * Yy;  Yw[3] = s3 * Z;
    Yw[4] = s15 * X * Yy;
    Yw[5] = s15 * Yy * Z;
    Yw[6] = 0.5f * s5 * (3.0f * Z * Z - 1.0f);
    Yw[7] = s15 * X * Z;
    Yw[8] = 0.5f * s15 * (X * X - Yy * Yy);
    Yw[9] = s70_4 * Yy * (3.0f * X * X - Yy * Yy);
    Yw[10] = s105 * X * Yy * Z;
    Yw[11] = s42_4 * Yy * (5.0f * Z * Z - 1.0f);
    Yw[12] = s7_2 * Z * (5.0f * Z * Z - 3.0f);
    Yw[13] = s42_4 * X * (5.0f * Z * Z - 1.0f);
    Yw[14] = s105_2 * Z * (X * X - Yy * Yy);
    Yw[15] = s70_4 * X * (X * X - 3.0f * Yy * Yy);

    // Bessel radial * poly cutoff (p=5)
    float u = r * 0.2f;
    float u2 = u * u, u4 = u2 * u2, u5 = u4 * u;
    float env = 1.0f - 21.0f * u5 + 35.0f * u5 * u - 15.0f * u5 * u2;
    float pref = 0.6324555320336759f / r;  // sqrt(2/RMAX)
#pragma unroll
    for (int b = 0; b < 8; ++b)
      sm.act0[w][lane][b] =
          (_Float16)(pref * __sinf((float)(b + 1) * 3.14159265358979f * u) * env);
#pragma unroll
    for (int b = 8; b < 32; ++b) sm.act0[w][lane][b] = (_Float16)0.0f;
    sm.rcv[w][lane] = eidx[Ee + e];  // receiver row
  }
  wait_async0();  // sender features resident in LDS
  __syncthreads();

  // Hoist receiver base pointers out of the scatter loops (loop-invariant).
  float* aggp[8];
#pragma unroll
  for (int rr = 0; rr < 8; ++rr)
    aggp[rr] = agg + (size_t)sm.rcv[w][rr + 8 * hh] * 2048;

  // ---- layer 1: [16x32(pad)] @ [32x64] ----
  {
    v16h a = frag_a_h(&sm.act0[w][0][0], 64, 0, lane);
#pragma unroll
    for (int t = 0; t < 4; ++t) {
      v16h b = frag_b_h(Wr1h, 64, t * 16, 0, lane);
      v8f acc = {};
      acc = wmma32(a, b, acc);
#pragma unroll
      for (int rr = 0; rr < 8; ++rr)
        sm.act1[w][rr + 8 * hh][t * 16 + nn] = (_Float16)silu(acc[rr]);
    }
  }
  __syncthreads();

  // ---- layer 2: [16x64] @ [64x64] ----
  {
    v16h a0 = frag_a_h(&sm.act1[w][0][0], 64, 0, lane);
    v16h a1 = frag_a_h(&sm.act1[w][0][0], 64, 32, lane);
#pragma unroll
    for (int t = 0; t < 4; ++t) {
      v8f acc = {};
      acc = wmma32(a0, frag_b_h(Wr2h, 64, t * 16, 0, lane), acc);
      acc = wmma32(a1, frag_b_h(Wr2h, 64, t * 16, 32, lane), acc);
#pragma unroll
      for (int rr = 0; rr < 8; ++rr)
        sm.act0[w][rr + 8 * hh][t * 16 + nn] = (_Float16)silu(acc[rr]);
    }
  }
  __syncthreads();

  // ---- layer 3: [16x64] @ [64x64] ----
  {
    v16h a0 = frag_a_h(&sm.act0[w][0][0], 64, 0, lane);
    v16h a1 = frag_a_h(&sm.act0[w][0][0], 64, 32, lane);
#pragma unroll
    for (int t = 0; t < 4; ++t) {
      v8f acc = {};
      acc = wmma32(a0, frag_b_h(Wr3h, 64, t * 16, 0, lane), acc);
      acc = wmma32(a1, frag_b_h(Wr3h, 64, t * 16, 32, lane), acc);
#pragma unroll
      for (int rr = 0; rr < 8; ++rr)
        sm.act1[w][rr + 8 * hh][t * 16 + nn] = (_Float16)silu(acc[rr]);
    }
  }
  __syncthreads();

  // ---- layer 4: [16x64] @ [64x512] fused with msg scatter ----
  {
    v16h a0 = frag_a_h(&sm.act1[w][0][0], 64, 0, lane);
    v16h a1 = frag_a_h(&sm.act1[w][0][0], 64, 32, lane);
    for (int t = 0; t < 32; ++t) {
      const int l = t >> 3;              // which angular block of Rw
      const int c = (t & 7) * 16 + nn;   // channel within C
      v8f acc = {};
      acc = wmma32(a0, frag_b_h(Wr4h, 512, t * 16, 0, lane), acc);
      acc = wmma32(a1, frag_b_h(Wr4h, 512, t * 16, 32, lane), acc);
      float hv[8];
#pragma unroll
      for (int rr = 0; rr < 8; ++rr) hv[rr] = sm.hs[w][rr + 8 * hh][c];
      const int mstart = l * l;        // LMAP block starts: 0,1,4,9
      const int mcnt = 2 * l + 1;      // counts: 1,3,5,7
      for (int mi = 0; mi < mcnt; ++mi) {
        const int m = mstart + mi;
#pragma unroll
        for (int rr = 0; rr < 8; ++rr) {
          const int er = rr + 8 * hh;
          float val = acc[rr] * sm.Yt[w][er][m] * hv[rr] * 0.0625f;  // /AVG
          atomicAdd(aggp[rr] + m * 128 + c, val);
        }
      }
    }
  }
}

// ---------------------------------------------------------------------------
// A[n,m,:] = agg[n,m,:] @ W_out[LMAP[m]]  — in-place over agg (each wave
// reads its 16-node x 1-msh rows fully before storing; blocks are disjoint).
// ---------------------------------------------------------------------------
__global__ __launch_bounds__(32) void k_gemm_A(float* agg,
                                               const _Float16* __restrict__ Wouth) {
  const int lane = threadIdx.x;
  const int n0t = blockIdx.x * 16;
  const int msh = blockIdx.y;
  const int l = (msh == 0) ? 0 : (msh < 4) ? 1 : (msh < 9) ? 2 : 3;
  const float* Arow = agg + (size_t)n0t * 2048 + msh * 128;
  const _Float16* B = Wouth + (size_t)l * 128 * 128;
  v8f zero = {};
  v8f acc[8];
#pragma unroll
  for (int t = 0; t < 8; ++t) acc[t] = zero;
#pragma unroll
  for (int kk = 0; kk < 128; kk += 32) {
    v16h a = frag_a_f(Arow, 2048, kk, lane);
#pragma unroll
    for (int t = 0; t < 8; ++t) {
      v16h b = frag_b_h(B, 128, t * 16, kk, lane);
      acc[t] = wmma32(a, b, acc[t]);
    }
  }
  const int hh = lane >> 4, nn = lane & 15;
#pragma unroll
  for (int t = 0; t < 8; ++t)
#pragma unroll
    for (int rr = 0; rr < 8; ++rr)
      agg[(size_t)(n0t + rr + 8 * hh) * 2048 + msh * 128 + t * 16 + nn] =
          acc[t][rr];
}

// ---------------------------------------------------------------------------
// Per-node nonlinear contraction: s, |A1|^2, |A2|^2, |A3|^2 -> B0/coef bases,
// p0 = sum_b (Wp0[elem] * B0),  p1 = A1 * sum_b (Wp1[elem] * coef).
// ---------------------------------------------------------------------------
__global__ __launch_bounds__(128) void k_nonlin(
    const float* __restrict__ A, const int* __restrict__ elem,
    const float* __restrict__ Wp0, const float* __restrict__ Wp1,
    _Float16* __restrict__ p0h, _Float16* __restrict__ p1h) {
  const int n = blockIdx.x, c = threadIdx.x;
  const float* An = A + (size_t)n * 2048;
  float s = An[c];
  float a1x = An[128 + c], a1y = An[256 + c], a1z = An[384 + c];
  float n1 = a1x * a1x + a1y * a1y + a1z * a1z;
  float n2 = 0.0f, n3 = 0.0f;
#pragma unroll
  for (int m = 4; m < 9; ++m) { float v = An[m * 128 + c]; n2 += v * v; }
#pragma unroll
  for (int m = 9; m < 16; ++m) { float v = An[m * 128 + c]; n3 += v * v; }
  const int e = elem[n];
  const float* w0 = Wp0 + (size_t)e * 9 * 128 + c;
  float B0[9] = {s, s * s, s * s * s, n1, n2, n3, s * n1, s * n2, s * n3};
  float p0 = 0.0f;
#pragma unroll
  for (int b = 0; b < 9; ++b) p0 += w0[b * 128] * B0[b];
  const float* w1 = Wp1 + (size_t)e * 6 * 128 + c;
  float coef[6] = {1.0f, s, s * s, n1, n2, n3};
  float q = 0.0f;
#pragma unroll
  for (int b = 0; b < 6; ++b) q += w1[b * 128] * coef[b];
  p0h[(size_t)n * 128 + c] = (_Float16)p0;
  p1h[((size_t)n * 3 + 0) * 128 + c] = (_Float16)(a1x * q);
  p1h[((size_t)n * 3 + 1) * 128 + c] = (_Float16)(a1y * q);
  p1h[((size_t)n * 3 + 2) * 128 + c] = (_Float16)(a1z * q);
}

// ---------------------------------------------------------------------------
// o0 = p0 @ Wlin0 + sc0  -> out[:, 0, :]
// ---------------------------------------------------------------------------
__global__ __launch_bounds__(32) void k_gemm_o0(
    const _Float16* __restrict__ p0h, const _Float16* __restrict__ Wl0h,
    const float* __restrict__ sc0, float* __restrict__ out) {
  const int lane = threadIdx.x;
  const size_t m0 = (size_t)blockIdx.x * 16;
  const int n0 = blockIdx.y * 16;
  v8f acc = {};
#pragma unroll
  for (int kk = 0; kk < 128; kk += 32) {
    v16h a = frag_a_h(p0h + m0 * 128, 128, kk, lane);
    v16h b = frag_b_h(Wl0h, 128, n0, kk, lane);
    acc = wmma32(a, b, acc);
  }
  const int hh = lane >> 4, nn = lane & 15;
#pragma unroll
  for (int rr = 0; rr < 8; ++rr) {
    size_t row = m0 + rr + 8 * hh;
    out[row * 512 + n0 + nn] = acc[rr] + sc0[row * 128 + n0 + nn];
  }
}

// ---------------------------------------------------------------------------
// o1 = p1 @ Wlin1  -> out[:, 1:4, :]   (rows are n*3+mvec)
// ---------------------------------------------------------------------------
__global__ __launch_bounds__(32) void k_gemm_o1(
    const _Float16* __restrict__ p1h, const _Float16* __restrict__ Wl1h,
    float* __restrict__ out) {
  const int lane = threadIdx.x;
  const size_t m0 = (size_t)blockIdx.x * 16;
  const int n0 = blockIdx.y * 16;
  v8f acc = {};
#pragma unroll
  for (int kk = 0; kk < 128; kk += 32) {
    v16h a = frag_a_h(p1h + m0 * 128, 128, kk, lane);
    v16h b = frag_b_h(Wl1h, 128, n0, kk, lane);
    acc = wmma32(a, b, acc);
  }
  const int hh = lane >> 4, nn = lane & 15;
#pragma unroll
  for (int rr = 0; rr < 8; ++rr) {
    size_t row = m0 + rr + 8 * hh;
    size_t node = row / 3;
    int mv = (int)(row % 3);
    out[node * 512 + (size_t)(mv + 1) * 128 + n0 + nn] = acc[rr];
  }
}

// ---------------------------------------------------------------------------
extern "C" void kernel_launch(void* const* d_in, const int* in_sizes, int n_in,
                              void* d_out, int out_size, void* d_ws,
                              size_t ws_size, hipStream_t stream) {
  const float* x     = (const float*)d_in[0];
  const float* evec  = (const float*)d_in[1];
  const float* elen  = (const float*)d_in[2];
  const int*   eidx  = (const int*)d_in[3];
  const float* Wemb  = (const float*)d_in[4];
  const float* Wskip = (const float*)d_in[5];
  const float* Wup   = (const float*)d_in[6];
  const float* Wr1   = (const float*)d_in[7];
  const float* Wr2   = (const float*)d_in[8];
  const float* Wr3   = (const float*)d_in[9];
  const float* Wr4   = (const float*)d_in[10];
  const float* Wout  = (const float*)d_in[11];
  const float* Wp0   = (const float*)d_in[12];
  const float* Wp1   = (const float*)d_in[13];
  const float* Wlin0 = (const float*)d_in[14];
  const float* Wlin1 = (const float*)d_in[15];
  float* out = (float*)d_out;

  char* base = (char*)d_ws;
  size_t off = 0;
  auto alloc = [&](size_t bytes) -> void* {
    void* p = base + off;
    off += (bytes + 255) & ~(size_t)255;
    return p;
  };
  float*    agg   = (float*)alloc((size_t)Nn * 2048 * 4);  // reused as A
  float*    hfeat = (float*)alloc((size_t)Nn * 128 * 4);
  float*    sc0   = (float*)alloc((size_t)Nn * 128 * 4);
  _Float16* h0h   = (_Float16*)alloc((size_t)Nn * 128 * 2);
  _Float16* p0h   = (_Float16*)alloc((size_t)Nn * 128 * 2);
  _Float16* p1h   = (_Float16*)alloc((size_t)Nn * 3 * 128 * 2);
  int*      elem  = (int*)alloc((size_t)Nn * 4);
  _Float16* Wr1h  = (_Float16*)alloc(32 * 64 * 2);
  _Float16* Wr2h  = (_Float16*)alloc(64 * 64 * 2);
  _Float16* Wr3h  = (_Float16*)alloc(64 * 64 * 2);
  _Float16* Wr4h  = (_Float16*)alloc(64 * 512 * 2);
  _Float16* Wuph  = (_Float16*)alloc(128 * 128 * 2);
  _Float16* Wouth = (_Float16*)alloc(4 * 128 * 128 * 2);
  _Float16* Wl0h  = (_Float16*)alloc(128 * 128 * 2);
  _Float16* Wl1h  = (_Float16*)alloc(128 * 128 * 2);

  k_zero<<<(Nn * 2048 + 255) / 256, 256, 0, stream>>>(agg, (size_t)Nn * 2048);
  k_wr1pad<<<8, 256, 0, stream>>>(Wr1, Wr1h);
  k_cvt<<<16, 256, 0, stream>>>(Wr2, Wr2h, 64 * 64);
  k_cvt<<<16, 256, 0, stream>>>(Wr3, Wr3h, 64 * 64);
  k_cvt<<<128, 256, 0, stream>>>(Wr4, Wr4h, 64 * 512);
  k_cvt<<<64, 256, 0, stream>>>(Wup, Wuph, 128 * 128);
  k_cvt<<<256, 256, 0, stream>>>(Wout, Wouth, 4 * 128 * 128);
  k_cvt<<<64, 256, 0, stream>>>(Wlin0, Wl0h, 128 * 128);
  k_cvt<<<64, 256, 0, stream>>>(Wlin1, Wl1h, 128 * 128);

  k_node_init<<<Nn, 128, 0, stream>>>(x, Wemb, Wskip, elem, h0h, sc0);
  k_gemm_h<<<dim3(Nn / 16, 8), 32, 0, stream>>>(h0h, Wuph, hfeat);
  k_edge<<<Ee / 64, 128, 0, stream>>>(evec, elen, eidx, hfeat, Wr1h, Wr2h,
                                      Wr3h, Wr4h, agg);
  k_gemm_A<<<dim3(Nn / 16, 16), 32, 0, stream>>>(agg, Wouth);
  k_nonlin<<<Nn, 128, 0, stream>>>(agg, elem, Wp0, Wp1, p0h, p1h);
  k_gemm_o0<<<dim3(Nn / 16, 8), 32, 0, stream>>>(p0h, Wl0h, sc0, out);
  k_gemm_o1<<<dim3(Nn * 3 / 16, 8), 32, 0, stream>>>(p1h, Wl1h, out);
}
